// GPSPolicy_22643067584999
// MI455X (gfx1250) — compile-verified, matrix operand order checked by reading, not run on hardware
//
#include <hip/hip_runtime.h>
#include <hip/hip_bf16.h>

// ---- problem constants (from reference) ----
#define N_NODES 16384
#define G_GRAPHS 64
#define NPG 256        // nodes per graph
#define H_ 256
#define E_EDGES 262144
#define ND_ 128
#define ED_ 16
#define PE_ 32
#define RW_ 20
#define A_ 16
#define L_ 4
#define NH_ 8
#define DH_ 32

typedef __attribute__((ext_vector_type(16))) _Float16 v16h;
typedef __attribute__((ext_vector_type(8)))  float    v8f;

union Frag16 { v16h v; _Float16 e[16]; };
union Frag8  { v8f  v; float     e[8];  };

// ============================================================
// WMMA GEMM: C[M,N] = act(A[M,K] @ B[K,N] + bias)
// A,B row-major f16; C row-major f32. M%128==0, N%128==0, K%32==0.
// Block = 256 threads = 8 waves (4x2); block tile 128x128;
// wave tile 32x64 (2x4 WMMA fragments -> 8 v_wmma per K-step).
// B is staged TRANSPOSED in LDS so both A and B fragments are two
// contiguous 8xf16 runs per lane -> ds_load_b128 pairs.
// ============================================================
template<int RELU>
__global__ __launch_bounds__(256) void wmma_gemm_kernel(
    const _Float16* __restrict__ A, const _Float16* __restrict__ B,
    const float* __restrict__ bias, float* __restrict__ C,
    int M, int Nn, int K)
{
    __shared__ __align__(16) _Float16 As[128 * 40];   // 128 rows x 32 K (+8 pad)
    __shared__ __align__(16) _Float16 Bt[128 * 40];   // 128 cols x 32 K (+8 pad), transposed

    const int tid  = threadIdx.x;
    const int lane = tid & 31;
    const int wave = tid >> 5;
    const int wm   = wave >> 1;          // 0..3 (M direction, 32 rows each)
    const int wn   = wave & 1;           // 0..1 (N direction, 64 cols each)
    const int l15  = lane & 15;
    const int hlf  = lane >> 4;
    const int mBase = blockIdx.y * 128;
    const int nBase = blockIdx.x * 128;

    Frag8 acc[2][4];
#pragma unroll
    for (int mt = 0; mt < 2; ++mt)
#pragma unroll
        for (int nt = 0; nt < 4; ++nt)
#pragma unroll
            for (int i = 0; i < 8; ++i) acc[mt][nt].e[i] = 0.f;

    // staging maps
    const int aRow = tid >> 1;            // 0..127
    const int aCol = (tid & 1) * 16;      // 0 / 16
    const int bRow = tid >> 3;            // 0..31 (K)
    const int bCol = (tid & 7) * 16;      // 0..112 (N, 16 wide)

    for (int k0 = 0; k0 < K; k0 += 32) {
        // stage A tile 128x32 (vector copies)
        const _Float16* ag = A + (size_t)(mBase + aRow) * K + k0 + aCol;
        *(uint4*)(&As[aRow * 40 + aCol])     = *(const uint4*)(ag);
        *(uint4*)(&As[aRow * 40 + aCol + 8]) = *(const uint4*)(ag + 8);
        // stage B tile 32x128 transposed: Bt[col*40 + k]
        {
            const _Float16* bg = B + (size_t)(k0 + bRow) * Nn + nBase + bCol;
            _Float16 tmp[16];
            *(uint4*)(&tmp[0]) = *(const uint4*)(bg);
            *(uint4*)(&tmp[8]) = *(const uint4*)(bg + 8);
#pragma unroll
            for (int j = 0; j < 16; ++j) Bt[(bCol + j) * 40 + bRow] = tmp[j];
        }
        __syncthreads();

        // fragments (ISA 7.12.2 16-bit 16x32 layout): per lane two contiguous
        // 8xf16 runs at K = hlf*8.. and K = 16+hlf*8..
        Frag16 af[2], bf[4];
#pragma unroll
        for (int mt = 0; mt < 2; ++mt) {
            const _Float16* ap = &As[(wm * 32 + mt * 16 + l15) * 40 + hlf * 8];
            *(uint4*)(&af[mt].e[0]) = *(const uint4*)(ap);
            *(uint4*)(&af[mt].e[8]) = *(const uint4*)(ap + 16);
        }
#pragma unroll
        for (int nt = 0; nt < 4; ++nt) {
            const _Float16* bp = &Bt[(wn * 64 + nt * 16 + l15) * 40 + hlf * 8];
            *(uint4*)(&bf[nt].e[0]) = *(const uint4*)(bp);
            *(uint4*)(&bf[nt].e[8]) = *(const uint4*)(bp + 16);
        }
#pragma unroll
        for (int mt = 0; mt < 2; ++mt)
#pragma unroll
            for (int nt = 0; nt < 4; ++nt)
                acc[mt][nt].v = __builtin_amdgcn_wmma_f32_16x16x32_f16(
                    false, af[mt].v, false, bf[nt].v, (short)0, acc[mt][nt].v, false, false);
        __syncthreads();
    }

    // epilogue: C/D layout — lane: N = lane%16; VGPR v: M = v + 8*(lane>=16)
#pragma unroll
    for (int mt = 0; mt < 2; ++mt) {
#pragma unroll
        for (int nt = 0; nt < 4; ++nt) {
            int col = nBase + wn * 64 + nt * 16 + l15;
            float bv = bias[col];
#pragma unroll
            for (int vv = 0; vv < 8; ++vv) {
                int row = mBase + wm * 32 + mt * 16 + vv + hlf * 8;
                float val = acc[mt][nt].e[vv] + bv;
                if (RELU) val = fmaxf(val, 0.f);
                C[(size_t)row * Nn + col] = val;
            }
        }
    }
}

// ============================================================
// elementwise helpers (all sizes are multiples of 256)
// ============================================================
__global__ void cvt_kernel(_Float16* __restrict__ dst, const float* __restrict__ src) {
    size_t i = (size_t)blockIdx.x * 256 + threadIdx.x;
    dst[i] = (_Float16)src[i];
}
__global__ void addcvt_kernel(_Float16* __restrict__ dst, const float* __restrict__ a,
                              const float* __restrict__ b) {
    size_t i = (size_t)blockIdx.x * 256 + threadIdx.x;
    dst[i] = (_Float16)(a[i] + b[i]);
}
__global__ void vadd_kernel(float* __restrict__ dst, const float* __restrict__ a,
                            const float* __restrict__ b) {
    size_t i = (size_t)blockIdx.x * 256 + threadIdx.x;
    dst[i] = a[i] + b[i];
}
__global__ void vzero_kernel(float* __restrict__ dst) {
    size_t i = (size_t)blockIdx.x * 256 + threadIdx.x;
    dst[i] = 0.f;
}

// ============================================================
// build concat([xproj, depth_emb[nd], rwse@pe_w+pe_b]) as f16 [N,544]
// ============================================================
__global__ void pack_concat_kernel(const float* __restrict__ xproj,
                                   const float* __restrict__ demb,
                                   const int* __restrict__ ndep,
                                   const float* __restrict__ rwse,
                                   const float* __restrict__ pe_w,
                                   const float* __restrict__ pe_b,
                                   _Float16* __restrict__ out) {
    size_t i = (size_t)blockIdx.x * 256 + threadIdx.x;   // over N*544
    int n = (int)(i / 544);
    int c = (int)(i % 544);
    float v;
    if (c < 256)      v = xproj[(size_t)n * 256 + c];
    else if (c < 512) v = demb[(size_t)ndep[n] * 256 + (c - 256)];
    else {
        int pc = c - 512;
        v = pe_b[pc];
#pragma unroll
        for (int k = 0; k < RW_; ++k) v += rwse[(size_t)n * RW_ + k] * pe_w[k * PE_ + pc];
    }
    out[i] = (_Float16)v;
}

// ============================================================
// GINE message + scatter: aggr[dst] += relu(h[src] + edge_attr@efw + efb)
// edgefeat projection recomputed on the fly (K=16) from LDS-staged weights.
// 32 lanes per edge, 8 cols per lane; 64 edges per block.
// ============================================================
__global__ __launch_bounds__(256) void gine_kernel(
    const float* __restrict__ h, const float* __restrict__ edge_attr,
    const int* __restrict__ edge_index, const float* __restrict__ efw,
    const float* __restrict__ efb, float* __restrict__ aggr)
{
    __shared__ float sW[ED_ * H_];
    __shared__ float sB[H_];
    for (int i = threadIdx.x; i < ED_ * H_; i += 256) sW[i] = efw[i];
    if (threadIdx.x < H_) sB[threadIdx.x] = efb[threadIdx.x];
    __syncthreads();

    const int lane = threadIdx.x & 31;
    const int sub  = threadIdx.x >> 5;
    const int c0   = lane * 8;
    const int eEnd = (blockIdx.x + 1) * 64;
    for (int e = blockIdx.x * 64 + sub; e < eEnd; e += 8) {
        int src = edge_index[e];
        int dst = edge_index[E_EDGES + e];
        float ea[ED_];
#pragma unroll
        for (int k = 0; k < ED_; ++k) ea[k] = edge_attr[(size_t)e * ED_ + k];
        float m[8];
#pragma unroll
        for (int j = 0; j < 8; ++j) m[j] = sB[c0 + j];
#pragma unroll
        for (int k = 0; k < ED_; ++k) {
            float wv = ea[k];
#pragma unroll
            for (int j = 0; j < 8; ++j) m[j] += wv * sW[k * H_ + c0 + j];
        }
        const float* hs = h + (size_t)src * H_ + c0;
        float* ap = aggr + (size_t)dst * H_ + c0;
#pragma unroll
        for (int j = 0; j < 8; ++j) {
            float v = hs[j] + m[j];
            v = fmaxf(v, 0.f);
            atomicAdd(&ap[j], v);
        }
    }
}

// ============================================================
// BatchNorm (training mode, biased variance) stats: one block per column
// ============================================================
__global__ __launch_bounds__(256) void colstats_kernel(const float* __restrict__ x,
                                                       float* __restrict__ mean,
                                                       float* __restrict__ rstd) {
    __shared__ float ssum[256], ssq[256];
    int c = blockIdx.x;
    float s = 0.f, q = 0.f;
    for (int r = threadIdx.x; r < N_NODES; r += 256) {
        float v = x[(size_t)r * H_ + c];
        s += v; q += v * v;
    }
    ssum[threadIdx.x] = s; ssq[threadIdx.x] = q;
    __syncthreads();
    for (int st = 128; st > 0; st >>= 1) {
        if (threadIdx.x < st) {
            ssum[threadIdx.x] += ssum[threadIdx.x + st];
            ssq[threadIdx.x]  += ssq[threadIdx.x + st];
        }
        __syncthreads();
    }
    if (threadIdx.x == 0) {
        float m = ssum[0] * (1.f / N_NODES);
        float v = ssq[0] * (1.f / N_NODES) - m * m;
        mean[c] = m;
        rstd[c] = rsqrtf(v + 1e-5f);
    }
}

__global__ void bn_apply_kernel(float* __restrict__ dst, const float* __restrict__ src,
                                const float* __restrict__ mean, const float* __restrict__ rstd,
                                const float* __restrict__ g, const float* __restrict__ b) {
    size_t i = (size_t)blockIdx.x * 256 + threadIdx.x;
    int c = (int)(i & (H_ - 1));
    dst[i] = (src[i] - mean[c]) * rstd[c] * g[c] + b[c];
}

// ============================================================
// split qkv [N,768] f32 -> q/k/v f16 each laid out [G,NH,NP,DH]
// ============================================================
__global__ void qkv_split_kernel(const float* __restrict__ qkv,
                                 _Float16* __restrict__ q, _Float16* __restrict__ k,
                                 _Float16* __restrict__ v) {
    size_t i = (size_t)blockIdx.x * 256 + threadIdx.x;   // over N*256
    int n = (int)(i >> 8);
    int c = (int)(i & 255);
    int hd = c >> 5, d = c & 31;
    int g = n >> 8, p = n & 255;
    size_t o = (((size_t)(g * NH_ + hd)) * NPG + p) * DH_ + d;
    const float* row = qkv + (size_t)n * 768;
    q[o] = (_Float16)row[c];
    k[o] = (_Float16)row[256 + c];
    v[o] = (_Float16)row[512 + c];
}

// ============================================================
// attention: block per (graph, head); K/V staged in LDS; online softmax,
// one query row per thread. Output ao f16 [N, H].
// ============================================================
__global__ __launch_bounds__(256) void attn_kernel(const _Float16* __restrict__ q,
                                                   const _Float16* __restrict__ k,
                                                   const _Float16* __restrict__ v,
                                                   _Float16* __restrict__ ao) {
    __shared__ __align__(16) _Float16 Ks[NPG * DH_];
    __shared__ __align__(16) _Float16 Vs[NPG * DH_];
    int gh = blockIdx.x;               // 0..511
    int g = gh >> 3, hd = gh & 7;
    const _Float16* kb = k + (size_t)gh * NPG * DH_;
    const _Float16* vb = v + (size_t)gh * NPG * DH_;
    for (int i = threadIdx.x * 8; i < NPG * DH_; i += 256 * 8) {
        *(uint4*)(&Ks[i]) = *(const uint4*)(kb + i);
        *(uint4*)(&Vs[i]) = *(const uint4*)(vb + i);
    }
    __syncthreads();

    int p = threadIdx.x;
    float qr[DH_];
    const _Float16* qp = q + ((size_t)gh * NPG + p) * DH_;
#pragma unroll
    for (int d = 0; d < DH_; ++d) qr[d] = (float)qp[d];

    float acc[DH_];
#pragma unroll
    for (int d = 0; d < DH_; ++d) acc[d] = 0.f;
    float mmax = -3.4e38f, lsum = 0.f;
    const float scale = 0.17677669529663687f;   // 1/sqrt(32)

    for (int j = 0; j < NPG; ++j) {
        const _Float16* kr = &Ks[j * DH_];
        float s = 0.f;
#pragma unroll
        for (int d = 0; d < DH_; ++d) s += qr[d] * (float)kr[d];
        s *= scale;
        float mn = fmaxf(mmax, s);
        float corr = __expf(mmax - mn);
        float pj = __expf(s - mn);
        lsum = lsum * corr + pj;
        const _Float16* vr = &Vs[j * DH_];
#pragma unroll
        for (int d = 0; d < DH_; ++d) acc[d] = acc[d] * corr + pj * (float)vr[d];
        mmax = mn;
    }
    float inv = 1.f / lsum;
    int n = g * NPG + p;
#pragma unroll
    for (int d = 0; d < DH_; ++d) ao[(size_t)n * H_ + hd * DH_ + d] = (_Float16)(acc[d] * inv);
}

// ============================================================
// readout: pi=[h[cur], mask@Wc+bc]; logits, gumbel-argmax, logp, vpred
// one block per graph. out = logits[1024] | act[64] | logp[64] | vpred[64]
// ============================================================
__device__ __forceinline__ unsigned int pcg_hash(unsigned int x) {
    x = x * 747796405u + 2891336453u;
    unsigned int w = ((x >> ((x >> 28) + 4u)) ^ x) * 277803737u;
    return (w >> 22) ^ w;
}

__global__ __launch_bounds__(256) void readout_kernel(
    const float* __restrict__ h, const int* __restrict__ cur_idx,
    const int* __restrict__ masks,
    const float* __restrict__ cw, const float* __restrict__ cb,
    const float* __restrict__ aw, const float* __restrict__ ab,
    const float* __restrict__ crw, const float* __restrict__ crb,
    float* __restrict__ out)
{
    __shared__ float pi[2 * H_];
    __shared__ float lg[A_];
    int g = blockIdx.x;
    int n = cur_idx[g];
    int j = threadIdx.x;                 // 0..255
    pi[j] = h[(size_t)n * H_ + j];
    float s = cb[j];
#pragma unroll
    for (int a = 0; a < A_; ++a)
        if (masks[g * A_ + a]) s += cw[a * H_ + j];
    pi[H_ + j] = s;
    __syncthreads();

    if (threadIdx.x < A_) {
        int a = threadIdx.x;
        float acc = ab[a];
        for (int jj = 0; jj < 2 * H_; ++jj) acc += pi[jj] * aw[jj * A_ + a];
        lg[a] = masks[g * A_ + a] ? acc : -1e6f;
    } else if (threadIdx.x == A_) {
        float acc = crb[0];
        for (int jj = 0; jj < 2 * H_; ++jj) acc += pi[jj] * crw[jj];
        out[1024 + 128 + g] = acc;       // vpred
    }
    __syncthreads();

    if (threadIdx.x == 0) {
        float best = -3.4e38f; int bi = 0;
        float lmax = -3.4e38f;
        for (int a = 0; a < A_; ++a) {
            out[g * A_ + a] = lg[a];
            unsigned int hv = pcg_hash(42u * 65537u + (unsigned)(g * A_ + a));
            float u = (float)(hv >> 8) * (1.f / 16777216.f);
            u = fmaxf(u, 1e-6f);
            float val = lg[a] - logf(-logf(u));
            if (val > best) { best = val; bi = a; }
            lmax = fmaxf(lmax, lg[a]);
        }
        float se = 0.f;
        for (int a = 0; a < A_; ++a) se += expf(lg[a] - lmax);
        out[1024 + g] = (float)bi;                       // act
        out[1024 + 64 + g] = lg[bi] - (lmax + logf(se)); // logp
    }
}

// ============================================================
// host orchestration
// ============================================================
extern "C" void kernel_launch(void* const* d_in, const int* in_sizes, int n_in,
                              void* d_out, int out_size, void* d_ws, size_t ws_size,
                              hipStream_t stream)
{
    (void)in_sizes; (void)n_in; (void)out_size; (void)ws_size;
    const float* x         = (const float*)d_in[0];
    const float* rwse      = (const float*)d_in[1];
    const float* edge_attr = (const float*)d_in[2];
    const int*   ndep      = (const int*)d_in[3];
    const int*   eidx      = (const int*)d_in[4];
    const int*   cur_idx   = (const int*)d_in[5];
    const int*   masks     = (const int*)d_in[6];
    const float* np_w = (const float*)d_in[7];
    const float* np_b = (const float*)d_in[8];
    const float* demb = (const float*)d_in[9];
    const float* pe_w = (const float*)d_in[10];
    const float* pe_b = (const float*)d_in[11];
    const float* nf_w = (const float*)d_in[12];
    const float* nf_b = (const float*)d_in[13];
    const float* ef_w = (const float*)d_in[14];
    const float* ef_b = (const float*)d_in[15];
    const float* gin_w1 = (const float*)d_in[16];
    const float* gin_b1 = (const float*)d_in[17];
    const float* gin_w2 = (const float*)d_in[18];
    const float* gin_b2 = (const float*)d_in[19];
    const float* ai_w = (const float*)d_in[20];
    const float* ai_b = (const float*)d_in[21];
    const float* aout_w = (const float*)d_in[22];
    const float* aout_b = (const float*)d_in[23];
    const float* bn1l_g = (const float*)d_in[24];
    const float* bn1l_b = (const float*)d_in[25];
    const float* bn1a_g = (const float*)d_in[26];
    const float* bn1a_b = (const float*)d_in[27];
    const float* ff_w1 = (const float*)d_in[28];
    const float* ff_b1 = (const float*)d_in[29];
    const float* ff_w2 = (const float*)d_in[30];
    const float* ff_b2 = (const float*)d_in[31];
    const float* bn2_g = (const float*)d_in[32];
    const float* bn2_b = (const float*)d_in[33];
    const float* c_w = (const float*)d_in[34];
    const float* c_b = (const float*)d_in[35];
    const float* act_w = (const float*)d_in[36];
    const float* act_b = (const float*)d_in[37];
    const float* cr_w = (const float*)d_in[38];
    const float* cr_b = (const float*)d_in[39];
    float* out = (float*)d_out;

    // ---- workspace carve-out ----
    char* wsp = (char*)d_ws;
    size_t off = 0;
    auto alloc = [&](size_t bytes) -> void* {
        void* p = wsp + off;
        off += (bytes + 255) & ~(size_t)255;
        return p;
    };
    float* h    = (float*)alloc((size_t)N_NODES * H_ * 4);
    float* aggr = (float*)alloc((size_t)N_NODES * H_ * 4);
    float* hl   = (float*)alloc((size_t)N_NODES * H_ * 4);
    float* ha   = (float*)alloc((size_t)N_NODES * H_ * 4);
    float* t3   = (float*)alloc((size_t)N_NODES * 768 * 4);
    _Float16* h16   = (_Float16*)alloc((size_t)N_NODES * H_ * 2);
    _Float16* b16   = (_Float16*)alloc((size_t)N_NODES * 768 * 2);
    _Float16* qkv16 = (_Float16*)alloc((size_t)N_NODES * 768 * 2);
    _Float16* w16   = (_Float16*)alloc((size_t)2793472 * 2);
    float* mean = (float*)alloc(H_ * 4);
    float* rstd = (float*)alloc(H_ * 4);

    // f16 weight arena offsets (elements)
    _Float16* wNP = w16 + 0;           // 128*256
    _Float16* wNF = w16 + 32768;       // 544*256
    _Float16* wG1 = w16 + 172032;      // 4*256*256
    _Float16* wG2 = w16 + 434176;
    _Float16* wAI = w16 + 696320;      // 4*256*768
    _Float16* wAO = w16 + 1482752;
    _Float16* wF1 = w16 + 1744896;     // 4*256*512
    _Float16* wF2 = w16 + 2269184;     // 4*512*256

#define CVT(dst, src, n) cvt_kernel<<<(n)/256, 256, 0, stream>>>(dst, src)
#define GEMM(Ap, Bp, biasp, Cp, Mv, Nv, Kv, ACT) \
    wmma_gemm_kernel<ACT><<<dim3((Nv)/128, (Mv)/128), 256, 0, stream>>>(Ap, Bp, biasp, Cp, Mv, Nv, Kv)

    // ---- weight conversion (idempotent, every call) ----
    CVT(wNP, np_w, 32768);
    CVT(wNF, nf_w, 139264);
    CVT(wG1, gin_w1, 262144);
    CVT(wG2, gin_w2, 262144);
    CVT(wAI, ai_w, 786432);
    CVT(wAO, aout_w, 262144);
    CVT(wF1, ff_w1, 524288);
    CVT(wF2, ff_w2, 524288);

    // ---- input embedding ----
    CVT(b16, x, N_NODES * ND_);
    GEMM(b16, wNP, np_b, t3, N_NODES, H_, ND_, 0);                    // xproj
    pack_concat_kernel<<<(N_NODES * 544) / 256, 256, 0, stream>>>(t3, demb, ndep, rwse, pe_w, pe_b, b16);
    GEMM(b16, wNF, nf_b, h, N_NODES, H_, 544, 0);                     // h [N,H]

    const int NHv = N_NODES * H_;
    _Float16* q16 = qkv16;
    _Float16* k16 = qkv16 + (size_t)N_NODES * H_;
    _Float16* v16 = qkv16 + (size_t)2 * N_NODES * H_;

    for (int l = 0; l < L_; ++l) {
        CVT(h16, h, NHv);

        // --- GINE ---
        vzero_kernel<<<NHv / 256, 256, 0, stream>>>(aggr);
        gine_kernel<<<E_EDGES / 64, 256, 0, stream>>>(h, edge_attr, eidx, ef_w, ef_b, aggr);
        addcvt_kernel<<<NHv / 256, 256, 0, stream>>>(b16, h, aggr);
        GEMM(b16, wG1 + (size_t)l * 65536, gin_b1 + l * H_, t3, N_NODES, H_, H_, 1);
        CVT(b16, t3, NHv);
        GEMM(b16, wG2 + (size_t)l * 65536, gin_b2 + l * H_, hl, N_NODES, H_, H_, 0);
        vadd_kernel<<<NHv / 256, 256, 0, stream>>>(hl, hl, h);
        colstats_kernel<<<H_, 256, 0, stream>>>(hl, mean, rstd);
        bn_apply_kernel<<<NHv / 256, 256, 0, stream>>>(hl, hl, mean, rstd, bn1l_g + l * H_, bn1l_b + l * H_);

        // --- global MHA ---
        GEMM(h16, wAI + (size_t)l * 196608, ai_b + l * 768, t3, N_NODES, 768, H_, 0);
        qkv_split_kernel<<<NHv / 256, 256, 0, stream>>>(t3, q16, k16, v16);
        attn_kernel<<<G_GRAPHS * NH_, 256, 0, stream>>>(q16, k16, v16, b16);
        GEMM(b16, wAO + (size_t)l * 65536, aout_b + l * H_, ha, N_NODES, H_, H_, 0);
        vadd_kernel<<<NHv / 256, 256, 0, stream>>>(ha, ha, h);
        colstats_kernel<<<H_, 256, 0, stream>>>(ha, mean, rstd);
        bn_apply_kernel<<<NHv / 256, 256, 0, stream>>>(ha, ha, mean, rstd, bn1a_g + l * H_, bn1a_b + l * H_);

        // --- combine + FFN ---
        vadd_kernel<<<NHv / 256, 256, 0, stream>>>(h, hl, ha);
        CVT(h16, h, NHv);
        GEMM(h16, wF1 + (size_t)l * 131072, ff_b1 + l * 512, t3, N_NODES, 512, H_, 1);
        CVT(b16, t3, N_NODES * 512);
        GEMM(b16, wF2 + (size_t)l * 131072, ff_b2 + l * H_, hl, N_NODES, H_, 512, 0);
        vadd_kernel<<<NHv / 256, 256, 0, stream>>>(hl, hl, h);
        colstats_kernel<<<H_, 256, 0, stream>>>(hl, mean, rstd);
        bn_apply_kernel<<<NHv / 256, 256, 0, stream>>>(h, hl, mean, rstd, bn2_g + l * H_, bn2_b + l * H_);
    }

    // ---- readout ----
    readout_kernel<<<G_GRAPHS, 256, 0, stream>>>(h, cur_idx, masks, c_w, c_b, act_w, act_b, cr_w, cr_b, out);

#undef CVT
#undef GEMM
}
